// CrossRPEAttention_80900003987878
// MI455X (gfx1250) — matrix operand order repaired and checked by reading, not compiled, not run on hardware
//
#include <hip/hip_runtime.h>

// CrossRPEAttention for MI455X (gfx1250, wave32, WMMA bf16 16x16x32)
// B=4, N=1024, C=768, H=12, d=64
// Round 2: async global->LDS staging (ASYNCcnt) + double-buffered LDS pipeline.

#define H_HEADS 12
#define N_TOK   1024
#define C_DIM   768
#define D_HEAD  64
#define B_BATCH 4
#define M_ROWS  (B_BATCH * N_TOK)   // 4096
#define SCALE_F 0.125f              // d^-0.5 = 64^-0.5

typedef __attribute__((ext_vector_type(16))) __bf16 v16bf;
typedef __attribute__((ext_vector_type(8)))  float  v8f;

union BFrag { v16bf v; uint4 q[2]; };

__device__ __forceinline__ __bf16 f2bf(float f) {
    union { float f; unsigned u; } a; a.f = f;
    unsigned r = a.u + 0x7FFFu + ((a.u >> 16) & 1u);   // RNE
    union { unsigned short s; __bf16 b; } o; o.s = (unsigned short)(r >> 16);
    return o.b;
}
__device__ __forceinline__ float bf2f(__bf16 b) {
    union { unsigned short s; __bf16 b; } i; i.b = b;
    union { unsigned u; float f; } o; o.u = ((unsigned)i.s) << 16;
    return o.f;
}

// ---- CDNA5 async global->LDS copies (ASYNCcnt-tracked, bypass VGPRs) ----
// Generic->LDS address: flat LDS aperture maps addr[31:0] to the LDS offset.
__device__ __forceinline__ unsigned ldsaddr(const void* p) {
    return (unsigned)(unsigned long long)p;
}
__device__ __forceinline__ void async_b128(const void* g, void* l) {
    asm volatile("global_load_async_to_lds_b128 %0, %1, off"
                 :: "v"(ldsaddr(l)), "v"(g) : "memory");
}
__device__ __forceinline__ void async_b32(const void* g, void* l) {
    asm volatile("global_load_async_to_lds_b32 %0, %1, off"
                 :: "v"(ldsaddr(l)), "v"(g) : "memory");
}
__device__ __forceinline__ void wait_async0() {
    asm volatile("s_wait_asynccnt 0x0" ::: "memory");
}

// A-matrix fragment (16x32 bf16, ISA 7.12.2): lane L row=L&15,
// half-wave hi: elems 0..7 = K hi*8..hi*8+7, elems 8..15 = K 16+hi*8..
__device__ __forceinline__ v16bf ldfragA(const __bf16* t, int row, int kb, int stride, int lane) {
    int off = row * stride + kb + ((lane >> 4) << 3);
    BFrag f;
    f.q[0] = *reinterpret_cast<const uint4*>(t + off);
    f.q[1] = *reinterpret_cast<const uint4*>(t + off + 16);
    return f.v;
}
// B-matrix fragment (32x16 bf16): lane col=L&15, half hi: elems 0..7 = K hi*8.., elems 8..15 = K 16+hi*8..
__device__ __forceinline__ v16bf ldfragB(const __bf16* t, int row, int kb, int stride, int lane) {
    int off = row * stride + kb + ((lane >> 4) << 3);
    BFrag f;
    f.q[0] = *reinterpret_cast<const uint4*>(t + off);
    f.q[1] = *reinterpret_cast<const uint4*>(t + off + 16);
    return f.v;
}

__device__ __forceinline__ v8f wmma_bf16(v16bf a, v16bf b, v8f c) {
    return __builtin_amdgcn_wmma_f32_16x16x32_bf16(false, a, false, b, (short)0, c, false, false);
}

// ---------------- prep kernels ----------------

__global__ __launch_bounds__(256) void cvt_x_kernel(const float* __restrict__ x,
                                                    __bf16* __restrict__ xbf, int n) {
    int i = blockIdx.x * 256 + threadIdx.x;
    if (i < n) xbf[i] = f2bf(x[i]);
}

// w (k-major [k][n]) -> wT bf16 ([n][k]) for 4 weights
__global__ __launch_bounds__(256) void transpose_w_kernel(
    const float* __restrict__ w0, const float* __restrict__ w1,
    const float* __restrict__ w2, const float* __restrict__ w3,
    __bf16* __restrict__ t0, __bf16* __restrict__ t1,
    __bf16* __restrict__ t2, __bf16* __restrict__ t3) {
    __shared__ float tile[32][33];
    int z = blockIdx.z;
    const float* src = (z == 0) ? w0 : (z == 1) ? w1 : (z == 2) ? w2 : w3;
    __bf16* dst = (z == 0) ? t0 : (z == 1) ? t1 : (z == 2) ? t2 : t3;
    int tx = threadIdx.x, ty = threadIdx.y;
    int nBase = blockIdx.x * 32;   // n (col in src)
    int kBase = blockIdx.y * 32;   // k (row in src)
    for (int i = ty; i < 32; i += 8)
        tile[i][tx] = src[(size_t)(kBase + i) * C_DIM + nBase + tx];
    __syncthreads();
    for (int i = ty; i < 32; i += 8)
        dst[(size_t)(nBase + i) * C_DIM + kBase + tx] = f2bf(tile[tx][i]);
}

// ---------------- QKV GEMM ----------------
// 128x128 block tile, K-step 32, 8 waves (2x4), each wave 64x32 -> 4x2 WMMA tiles.
// Double-buffered LDS; next tile staged by async-to-LDS while WMMAs run.

#define GBM 128
#define GBN 128
#define GBK 32
#define GSTR 40   // padded LDS stride (bf16 elems): 80B, 16B-aligned, conflict-free

__global__ __launch_bounds__(256) void gemm_qkv_kernel(
    const __bf16* __restrict__ A,
    const __bf16* __restrict__ wqT, const __bf16* __restrict__ wkT,
    const __bf16* __restrict__ wvT,
    __bf16* __restrict__ qO, __bf16* __restrict__ kO, __bf16* __restrict__ vTO) {
    __shared__ __bf16 As[2 * GBM * GSTR];
    __shared__ __bf16 Bs[2 * GBN * GSTR];
    const int tid = threadIdx.x, lane = tid & 31, wid = tid >> 5;
    const int wm = wid >> 2, wn = wid & 3;
    const int m0 = blockIdx.x * GBM;
    const int mat = blockIdx.y / 6;
    const int n0 = (blockIdx.y % 6) * GBN;
    const __bf16* BT = (mat == 0) ? wqT : (mat == 1) ? wkT : wvT;

    v8f zero = {};
    v8f acc[4][2];
#pragma unroll
    for (int s = 0; s < 4; s++)
#pragma unroll
        for (int t = 0; t < 2; t++) acc[s][t] = zero;

    const int r = tid >> 1, seg = tid & 1;   // 128 rows x 2 x 16-elem segments
    const int ln = lane & 15;

    auto stage = [&](int buf, int k0) {
        const __bf16* sa = A + (size_t)(m0 + r) * C_DIM + k0 + seg * 16;
        __bf16* da = As + buf * GBM * GSTR + r * GSTR + seg * 16;
        async_b128(sa, da);
        async_b128(sa + 8, da + 8);
        const __bf16* sb = BT + (size_t)(n0 + r) * C_DIM + k0 + seg * 16;
        __bf16* db = Bs + buf * GBN * GSTR + r * GSTR + seg * 16;
        async_b128(sb, db);
        async_b128(sb + 8, db + 8);
    };

    stage(0, 0);
    wait_async0();
    __syncthreads();

    int cur = 0;
    for (int k0 = 0; k0 < C_DIM; k0 += GBK) {
        if (k0 + GBK < C_DIM) stage(cur ^ 1, k0 + GBK);
        if (k0 + 2 * GBK < C_DIM) {
            __builtin_prefetch(A + (size_t)(m0 + r) * C_DIM + k0 + 2 * GBK, 0, 0);
            __builtin_prefetch(BT + (size_t)(n0 + r) * C_DIM + k0 + 2 * GBK, 0, 0);
        }
        const __bf16* Ab = As + cur * GBM * GSTR;
        const __bf16* Bb = Bs + cur * GBN * GSTR;
        v16bf af[4], bfr[2];
#pragma unroll
        for (int s = 0; s < 4; s++) af[s] = ldfragA(Ab, wm * 64 + s * 16 + ln, 0, GSTR, lane);
#pragma unroll
        for (int t = 0; t < 2; t++) bfr[t] = ldfragB(Bb, wn * 32 + t * 16 + ln, 0, GSTR, lane);
#pragma unroll
        for (int s = 0; s < 4; s++)
#pragma unroll
            for (int t = 0; t < 2; t++) acc[s][t] = wmma_bf16(af[s], bfr[t], acc[s][t]);
        wait_async0();       // next tile landed in LDS (own wave's issues)
        __syncthreads();     // visible to all waves; current-buffer reads done
        cur ^= 1;
    }

    const int hi8 = (lane >> 4) << 3;
#pragma unroll
    for (int s = 0; s < 4; s++)
#pragma unroll
        for (int t = 0; t < 2; t++)
#pragma unroll
            for (int jv = 0; jv < 8; jv++) {
                int gm = m0 + wm * 64 + s * 16 + jv + hi8;   // row: b*1024 + tok
                int gc = n0 + wn * 32 + t * 16 + ln;         // col: h*64 + dd
                int b = gm >> 10, n = gm & 1023, h = gc >> 6, dd = gc & 63;
                __bf16 val = f2bf(acc[s][t][jv]);
                if (mat == 0)
                    qO[(((size_t)b * H_HEADS + h) * N_TOK + n) * D_HEAD + dd] = val;
                else if (mat == 1)
                    kO[(((size_t)b * H_HEADS + h) * N_TOK + n) * D_HEAD + dd] = val;
                else
                    vTO[(((size_t)b * H_HEADS + h) * D_HEAD + dd) * N_TOK + n] = val;
            }
}

// ---------------- RPE bias: bk = Q.rpe_k^T ; bq = scale*K.rpe_q^T ----------------

__global__ __launch_bounds__(256) void rpe_bias_kernel(
    const __bf16* __restrict__ qbf, const __bf16* __restrict__ kbf,
    const float* __restrict__ tabK, const float* __restrict__ tabQ,
    float* __restrict__ bk, float* __restrict__ bq) {
    int t = blockIdx.x * 256 + threadIdx.x;
    if (t >= B_BATCH * H_HEADS * N_TOK) return;
    const __bf16* qr = qbf + (size_t)t * D_HEAD;
    const __bf16* kr = kbf + (size_t)t * D_HEAD;
    float aK[5] = {0, 0, 0, 0, 0}, aQ[5] = {0, 0, 0, 0, 0};
    for (int d = 0; d < D_HEAD; d++) {
        float qv = bf2f(qr[d]), kv = bf2f(kr[d]);
#pragma unroll
        for (int u = 0; u < 5; u++) {
            aK[u] += qv * tabK[u * D_HEAD + d];
            aQ[u] += kv * tabQ[u * D_HEAD + d];
        }
    }
#pragma unroll
    for (int u = 0; u < 5; u++) {
        bk[(size_t)t * 5 + u] = aK[u];
        bq[(size_t)t * 5 + u] = aQ[u] * SCALE_F;   // scale baked in (rpe_q applied to k*scale)
    }
}

// ---------------- flash attention with RPE bucket biases ----------------
// grid: (N/128 query blocks, B*H). 8 waves x 16 query rows. Key chunks of 64,
// double-buffered via async-to-LDS. rp_bucket is symmetric (euclidean) -> one
// gather feeds both bk[i,.] and bq[j,.].

#define KSTR 72   // padded LDS stride: 144B, 16B-aligned, conflict-free for b128 frags

__global__ __launch_bounds__(256) void attn_kernel(
    const __bf16* __restrict__ qbf, const __bf16* __restrict__ kbf,
    const __bf16* __restrict__ vT,
    const float* __restrict__ bkg, const float* __restrict__ bqg,
    const int* __restrict__ rp, __bf16* __restrict__ attnout) {
    __shared__ __bf16 kt[2 * 64 * KSTR];        // [key][d]
    __shared__ __bf16 vt[2 * 64 * KSTR];        // [d][key]
    __shared__ __bf16 ps[8 * 16 * KSTR];        // per-wave P staging (D-layout -> A-layout)
    __shared__ float  bk_s[128 * 5];
    __shared__ float  bq_s[2 * 64 * 5];

    const int tid = threadIdx.x, lane = tid & 31, wid = tid >> 5;
    const int q0 = blockIdx.x * 128;
    const int bh = blockIdx.y;
    const int b = bh / H_HEADS, h = bh % H_HEADS;
    const size_t qkbase = (size_t)bh * N_TOK * D_HEAD;
    const int ln = lane & 15, hi8 = (lane >> 4) << 3;

    auto stageChunk = [&](int buf, int j0) {
        int r = tid >> 2, seg = tid & 3;
        const __bf16* s1 = kbf + qkbase + (size_t)(j0 + r) * D_HEAD + seg * 16;
        __bf16* d1 = kt + buf * 64 * KSTR + r * KSTR + seg * 16;
        async_b128(s1, d1);
        async_b128(s1 + 8, d1 + 8);
        const __bf16* s2 = vT + ((size_t)bh * D_HEAD + r) * N_TOK + j0 + seg * 16;
        __bf16* d2 = vt + buf * 64 * KSTR + r * KSTR + seg * 16;
        async_b128(s2, d2);
        async_b128(s2 + 8, d2 + 8);
        for (int i = tid; i < 64 * 5; i += 256)
            async_b32(bqg + ((size_t)bh * N_TOK + j0) * 5 + i, bq_s + buf * 64 * 5 + i);
    };

    // persistent Q fragments for this wave's 16 rows (A-layout, direct from global)
    v16bf qf[2];
    {
        const __bf16* qp = qbf + qkbase + (size_t)(q0 + wid * 16 + ln) * D_HEAD;
#pragma unroll
        for (int c = 0; c < 2; c++) {
            int base = c * 32 + hi8;
            BFrag f;
            f.q[0] = *(const uint4*)(qp + base);
            f.q[1] = *(const uint4*)(qp + base + 16);
            qf[c] = f.v;
        }
    }
    stageChunk(0, 0);
    for (int i = tid; i < 128 * 5; i += 256)
        async_b32(bkg + ((size_t)bh * N_TOK + q0) * 5 + i, bk_s + i);
    wait_async0();
    __syncthreads();

    v8f zero = {};
    v8f O[4]; float mr[8], lr[8];
#pragma unroll
    for (int i = 0; i < 4; i++) O[i] = zero;
#pragma unroll
    for (int i = 0; i < 8; i++) { mr[i] = -1e30f; lr[i] = 0.f; }

    int cur = 0;
    for (int j0 = 0; j0 < N_TOK; j0 += 64) {
        if (j0 + 64 < N_TOK) stageChunk(cur ^ 1, j0 + 64);
        const __bf16* ktb = kt + cur * 64 * KSTR;
        const __bf16* vtb = vt + cur * 64 * KSTR;
        const float*  bqb = bq_s + cur * 64 * 5;

        // S = Q K^T over 4 key subtiles
        v8f S[4];
#pragma unroll
        for (int t = 0; t < 4; t++) {
            v8f c = zero;
            c = wmma_bf16(qf[0], ldfragB(ktb, t * 16 + ln, 0, KSTR, lane), c);
            c = wmma_bf16(qf[1], ldfragB(ktb, t * 16 + ln, 32, KSTR, lane), c);
            S[t] = c;
        }
        // scale + bucket-gathered biases
#pragma unroll
        for (int t = 0; t < 4; t++)
#pragma unroll
            for (int jv = 0; jv < 8; jv++) {
                int gq = q0 + wid * 16 + jv + hi8;
                int gk = j0 + t * 16 + ln;
                int bu = rp[(size_t)gq * N_TOK + gk];
                S[t][jv] = S[t][jv] * SCALE_F
                         + bk_s[(wid * 16 + jv + hi8) * 5 + bu]
                         + bqb[(t * 16 + ln) * 5 + bu];
            }
        // online softmax (rows live across 16-lane half-waves per D-layout)
#pragma unroll
        for (int jv = 0; jv < 8; jv++) {
            float mx = fmaxf(fmaxf(S[0][jv], S[1][jv]), fmaxf(S[2][jv], S[3][jv]));
#pragma unroll
            for (int off = 8; off > 0; off >>= 1) mx = fmaxf(mx, __shfl_xor(mx, off, 32));
            float nm = fmaxf(mr[jv], mx);
            float al = __expf(mr[jv] - nm);
            float rs = 0.f;
#pragma unroll
            for (int t = 0; t < 4; t++) {
                float p = __expf(S[t][jv] - nm);
                ps[(wid * 16 + jv + hi8) * KSTR + t * 16 + ln] = f2bf(p);
                rs += p;
            }
#pragma unroll
            for (int off = 8; off > 0; off >>= 1) rs += __shfl_xor(rs, off, 32);
            lr[jv] = lr[jv] * al + rs;
            mr[jv] = nm;
#pragma unroll
            for (int ds = 0; ds < 4; ds++) O[ds][jv] *= al;
        }
        __syncthreads();   // P staging visible

        // O += P V
        v16bf pf[2];
#pragma unroll
        for (int c = 0; c < 2; c++)
            pf[c] = ldfragA(ps + wid * 16 * KSTR, ln, c * 32, KSTR, lane);
#pragma unroll
        for (int ds = 0; ds < 4; ds++) {
            O[ds] = wmma_bf16(pf[0], ldfragB(vtb, ds * 16 + ln, 0, KSTR, lane), O[ds]);
            O[ds] = wmma_bf16(pf[1], ldfragB(vtb, ds * 16 + ln, 32, KSTR, lane), O[ds]);
        }
        wait_async0();     // next chunk landed (own wave's async issues)
        __syncthreads();   // all waves' writes visible; current-buffer reads done
        cur ^= 1;
    }

    // normalize + store as bf16 (B,N,C) for the projection GEMM
#pragma unroll
    for (int jv = 0; jv < 8; jv++) {
        float inv = 1.f / lr[jv];
        int tok = q0 + wid * 16 + jv + hi8;
#pragma unroll
        for (int ds = 0; ds < 4; ds++)
            attnout[((size_t)(b * N_TOK + tok)) * C_DIM + h * D_HEAD + ds * 16 + ln] =
                f2bf(O[ds][jv] * inv);
    }
}

// ---------------- output projection GEMM (+bias, fp32 out) ----------------

__global__ __launch_bounds__(256) void gemm_proj_kernel(
    const __bf16* __restrict__ A, const __bf16* __restrict__ wpT,
    const float* __restrict__ bias, float* __restrict__ out) {
    __shared__ __bf16 As[2 * GBM * GSTR];
    __shared__ __bf16 Bs[2 * GBN * GSTR];
    const int tid = threadIdx.x, lane = tid & 31, wid = tid >> 5;
    const int wm = wid >> 2, wn = wid & 3;
    const int m0 = blockIdx.x * GBM;
    const int n0 = blockIdx.y * GBN;

    v8f zero = {};
    v8f acc[4][2];
#pragma unroll
    for (int s = 0; s < 4; s++)
#pragma unroll
        for (int t = 0; t < 2; t++) acc[s][t] = zero;

    const int r = tid >> 1, seg = tid & 1;
    const int ln = lane & 15;

    auto stage = [&](int buf, int k0) {
        const __bf16* sa = A + (size_t)(m0 + r) * C_DIM + k0 + seg * 16;
        __bf16* da = As + buf * GBM * GSTR + r * GSTR + seg * 16;
        async_b128(sa, da);
        async_b128(sa + 8, da + 8);
        const __bf16* sb = wpT + (size_t)(n0 + r) * C_DIM + k0 + seg * 16;
        __bf16* db = Bs + buf * GBN * GSTR + r * GSTR + seg * 16;
        async_b128(sb, db);
        async_b128(sb + 8, db + 8);
    };

    stage(0, 0);
    wait_async0();
    __syncthreads();

    int cur = 0;
    for (int k0 = 0; k0 < C_DIM; k0 += GBK) {
        if (k0 + GBK < C_DIM) stage(cur ^ 1, k0 + GBK);
        if (k0 + 2 * GBK < C_DIM) {
            __builtin_prefetch(A + (size_t)(m0 + r) * C_DIM + k0 + 2 * GBK, 0, 0);
            __builtin_prefetch(wpT + (size_t)(n0 + r) * C_DIM + k0 + 2 * GBK, 0, 0);
        }
        const __bf16* Ab = As + cur * GBM * GSTR;
        const __bf16* Bb = Bs + cur * GBN * GSTR;
        v16bf af[4], bfr[2];
#pragma unroll
        for (int s = 0; s < 4; s++) af[s] = ldfragA(Ab, wm * 64 + s * 16 + ln, 0, GSTR, lane);
#pragma unroll
        for (int t = 0; t < 2; t++) bfr[t] = ldfragB(Bb, wn * 32 + t * 16 + ln, 0, GSTR, lane);
#pragma unroll
        for (int s = 0; s < 4; s++)
#pragma unroll
            for (int t = 0; t < 2; t++) acc[s][t] = wmma_bf16(af[s], bfr[t], acc[s][t]);
        wait_async0();
        __syncthreads();
        cur ^= 1;
    }

    const int hi8 = (lane >> 4) << 3;
#pragma unroll
    for (int s = 0; s < 4; s++)
#pragma unroll
        for (int t = 0; t < 2; t++)
#pragma unroll
            for (int jv = 0; jv < 8; jv++) {
                int gm = m0 + wm * 64 + s * 16 + jv + hi8;
                int gc = n0 + wn * 32 + t * 16 + ln;
                out[(size_t)gm * C_DIM + gc] = acc[s][t][jv] + bias[gc];
            }
}

// ---------------- launcher ----------------

extern "C" void kernel_launch(void* const* d_in, const int* in_sizes, int n_in,
                              void* d_out, int out_size, void* d_ws, size_t ws_size,
                              hipStream_t stream) {
    const float* x   = (const float*)d_in[0];
    const float* wq  = (const float*)d_in[1];
    const float* wk  = (const float*)d_in[2];
    const float* wvw = (const float*)d_in[3];
    const float* wp  = (const float*)d_in[4];
    const float* pb  = (const float*)d_in[5];
    const float* tk  = (const float*)d_in[6];
    const float* tq  = (const float*)d_in[7];
    const int*   rp  = (const int*)d_in[8];
    float* out = (float*)d_out;

    char* ws = (char*)d_ws;
    size_t off = 0;
    auto alloc = [&](size_t bytes) -> void* {
        void* p = ws + off;
        off += (bytes + 255) & ~(size_t)255;
        return p;
    };
    __bf16* xbf  = (__bf16*)alloc((size_t)M_ROWS * C_DIM * 2);
    __bf16* wqT  = (__bf16*)alloc((size_t)C_DIM * C_DIM * 2);
    __bf16* wkT  = (__bf16*)alloc((size_t)C_DIM * C_DIM * 2);
    __bf16* wvT  = (__bf16*)alloc((size_t)C_DIM * C_DIM * 2);
    __bf16* wpT  = (__bf16*)alloc((size_t)C_DIM * C_DIM * 2);
    __bf16* qbf  = (__bf16*)alloc((size_t)B_BATCH * H_HEADS * N_TOK * D_HEAD * 2);
    __bf16* kbf  = (__bf16*)alloc((size_t)B_BATCH * H_HEADS * N_TOK * D_HEAD * 2);
    __bf16* vTbf = (__bf16*)alloc((size_t)B_BATCH * H_HEADS * N_TOK * D_HEAD * 2);
    __bf16* aout = (__bf16*)alloc((size_t)M_ROWS * C_DIM * 2);
    float*  bk   = (float*)alloc((size_t)B_BATCH * H_HEADS * N_TOK * 5 * 4);
    float*  bq   = (float*)alloc((size_t)B_BATCH * H_HEADS * N_TOK * 5 * 4);
    (void)in_sizes; (void)n_in; (void)out_size; (void)ws_size;

    int nx = M_ROWS * C_DIM;
    cvt_x_kernel<<<(nx + 255) / 256, 256, 0, stream>>>(x, xbf, nx);
    transpose_w_kernel<<<dim3(C_DIM / 32, C_DIM / 32, 4), dim3(32, 8), 0, stream>>>(
        wq, wk, wvw, wp, wqT, wkT, wvT, wpT);
    gemm_qkv_kernel<<<dim3(M_ROWS / GBM, 18), 256, 0, stream>>>(
        xbf, wqT, wkT, wvT, qbf, kbf, vTbf);
    rpe_bias_kernel<<<(B_BATCH * H_HEADS * N_TOK) / 256, 256, 0, stream>>>(
        qbf, kbf, tk, tq, bk, bq);
    attn_kernel<<<dim3(N_TOK / 128, B_BATCH * H_HEADS), 256, 0, stream>>>(
        qbf, kbf, vTbf, bk, bq, rp, aout);
    gemm_proj_kernel<<<dim3(M_ROWS / GBM, C_DIM / GBN), 256, 0, stream>>>(
        aout, wpT, pb, out);
}
